// RoPeMultiHeadAttn_67491116089407
// MI455X (gfx1250) — matrix operand order repaired and checked by reading, not compiled, not run on hardware
//
#include <hip/hip_runtime.h>
#include <hip/hip_bf16.h>

// ---------------------------------------------------------------------------
// RoPE multi-head attention forward for MI455X (gfx1250, wave32, WMMA).
// Pipeline:
//   1) cvt_x:        x f32 -> bf16                  (ws)
//   2) cvt_w_t:      W f32 [k][n] -> Wt bf16 [n][k] (ws, x4)
//   3) gemm_qkvo z=0..2:  Q/K/V = x@W + b, fused RoPE epilogue.
//   4) attn:         flash attention per (head, 128-query tile), bf16 WMMA
//   5) gemm_qkvo z=3: O = attnOut @ Wo + bo -> d_out
// Staging: per-lane GLOBAL_LOAD_ASYNC_TO_LDS_B128 (ASYNCcnt) for A/Q/K tiles,
// Tensor Data Mover descriptors (TENSORcnt) for the strided B / V^T tiles,
// all double-buffered so HBM/L2 latency overlaps the WMMA compute phase.
// ---------------------------------------------------------------------------

typedef __attribute__((ext_vector_type(16))) __bf16 v16bf;
typedef __attribute__((ext_vector_type(8)))  __bf16 v8bf;
typedef __attribute__((ext_vector_type(4)))  __bf16 v4bf;
typedef __attribute__((ext_vector_type(8)))  float  v8f;
typedef __attribute__((ext_vector_type(4)))  unsigned int u32x4;
typedef __attribute__((ext_vector_type(4)))  int  i32x4;
typedef __attribute__((ext_vector_type(8)))  int  i32x8;

static constexpr int QLEN  = 2048;
static constexpr int BSZ   = 4;
static constexpr int DM    = 1024;   // d_model
static constexpr int NH    = 16;
static constexpr int DH    = 64;
static constexpr int H     = NH * DH;          // 1024
static constexpr int ROWS  = QLEN * BSZ;       // 8192
static constexpr float SCALE = 0.125f;         // 1/sqrt(64)

#if defined(__has_builtin)
#if __has_builtin(__builtin_amdgcn_tensor_load_to_lds)
#define HAVE_TDM 1
#endif
#endif
#ifndef HAVE_TDM
#define HAVE_TDM 0
#endif

// ---- CDNA5 async global->LDS copy (CDNA5 ISA / 15.18.3) ------
// 16 bytes per lane, tracked by ASYNCcnt. LDS destination operand is the
// wave-relative LDS byte offset (low 32 bits of the generic shared pointer).
__device__ __forceinline__ void async_copy16(void* lds, const void* gaddr) {
  unsigned ldsOff = (unsigned)(unsigned long long)lds;
  asm volatile("global_load_async_to_lds_b128 %0, %1, off"
               :: "v"(ldsOff), "v"(gaddr) : "memory");
}
__device__ __forceinline__ void wait_async0() {
  asm volatile("s_wait_asynccnt 0x0" ::: "memory");
}

#if HAVE_TDM
// ---- Tensor Data Mover: 2D tile load, bf16 elements (ISA 08 §7/§8) --------
// Descriptor built per D# groups 0/1 (groups 2/3 zero => 2D tensor).
// dim0 = contiguous element dim; rows of the tile land densely in LDS.
__device__ __forceinline__ void tdm_load_2d(void* lds, const void* gaddr,
                                            unsigned tensorD0, unsigned tensorD1,
                                            unsigned tileD0,   unsigned tileD1,
                                            unsigned strideD0 /* elements */) {
  unsigned long long ga = (unsigned long long)gaddr;
  u32x4 g0;
  g0[0] = 1u;                                          // count=1, user mode
  g0[1] = (unsigned)(unsigned long long)lds;           // lds_addr (bytes)
  g0[2] = (unsigned)(ga & 0xffffffffu);                // global_addr[31:0]
  g0[3] = (unsigned)((ga >> 32) & 0x01ffffffu)         // global_addr[56:32]
          | 0x80000000u;                               // type=2 ("image")
  i32x8 g1;
  g1[0] = (int)(1u << 16);                             // data_size=1 (2B)
  g1[1] = (int)(tensorD0 << 16);                       // tensor_dim0[15:0]
  g1[2] = (int)((tensorD0 >> 16) | (tensorD1 << 16));  // dim0[31:16]|dim1[15:0]
  g1[3] = (int)((tensorD1 >> 16) | (tileD0 << 16));    // dim1[31:16]|tile_dim0
  g1[4] = (int)(tileD1 & 0xffffu);                     // tile_dim1 (tile_dim2=0)
  g1[5] = (int)strideD0;                               // tensor_dim0_stride
  g1[6] = 0;
  g1[7] = 0;
  i32x4 z4 = {0, 0, 0, 0};
#if defined(__clang_major__) && (__clang_major__ >= 23)
  i32x8 z8 = {0, 0, 0, 0, 0, 0, 0, 0};
  __builtin_amdgcn_tensor_load_to_lds(g0, g1, z4, z4, z8, 0);
#else
  __builtin_amdgcn_tensor_load_to_lds(g0, g1, z4, z4, 0);
#endif
}
__device__ __forceinline__ void wait_tensor0() {
#if __has_builtin(__builtin_amdgcn_s_wait_tensorcnt)
  __builtin_amdgcn_s_wait_tensorcnt(0);
#else
  asm volatile("s_wait_tensorcnt 0x0" ::: "memory");
#endif
}
#endif  // HAVE_TDM

// ---- WMMA fragment loaders (CDNA5 ISA 7.12.2, 16-bit A/B layouts) ----
__device__ __forceinline__ v16bf frag_a(const __bf16* row, int k0, int half) {
  v8bf lo = *(const v8bf*)(row + k0 + 8 * half);
  v8bf hi = *(const v8bf*)(row + k0 + 16 + 8 * half);
  v16bf r;
#pragma unroll
  for (int i = 0; i < 8; ++i) { r[i] = lo[i]; r[8 + i] = hi[i]; }
  return r;
}
__device__ __forceinline__ v16bf frag_b(const __bf16* rowT, int k0, int half) {
  v8bf lo = *(const v8bf*)(rowT + k0 + 8 * half);
  v8bf hi = *(const v8bf*)(rowT + k0 + 16 + 8 * half);
  v16bf r;
#pragma unroll
  for (int i = 0; i < 8; ++i) { r[i] = lo[i]; r[8 + i] = hi[i]; }
  return r;
}
__device__ __forceinline__ v8f wmma_bf16(v16bf a, v16bf b, v8f c) {
  return __builtin_amdgcn_wmma_f32_16x16x32_bf16(false, a, false, b,
                                                 (short)0, c, false, false);
}

// ---------------------------- conversion kernels ---------------------------
__global__ __launch_bounds__(256) void cvt_x(const float* __restrict__ x,
                                             __bf16* __restrict__ xb) {
  int i = (blockIdx.x * 256 + threadIdx.x) * 4;
  float4 v = *(const float4*)(x + i);
  v4bf o; o[0] = (__bf16)v.x; o[1] = (__bf16)v.y; o[2] = (__bf16)v.z; o[3] = (__bf16)v.w;
  *(v4bf*)(xb + i) = o;
}

__global__ __launch_bounds__(256) void cvt_w_t(const float* __restrict__ W,
                                               __bf16* __restrict__ Wt) {
  int idx = blockIdx.x * 256 + threadIdx.x;   // over 1024*1024
  int k = idx >> 10, n = idx & 1023;
  Wt[(size_t)n * 1024 + k] = (__bf16)W[idx];
}

// ------------------------------- GEMM kernel -------------------------------
struct GemmArgs {
  const __bf16* A;
  const __bf16* Wt[4];
  const float*  bias[4];
  const float*  fcos;
  const float*  fsin;
  float* out_k_f32;
  float* out_v_f32;
  float* out_o_f32;
  __bf16* qro;
  __bf16* kro;
  __bf16* vt;
  int modeBase;
};

static constexpr int BM = 128, BN = 128, BK = 64;

__global__ __launch_bounds__(256) void gemm_qkvo(GemmArgs args) {
  const int mode = args.modeBase + blockIdx.z;
  const __bf16* __restrict__ A  = args.A;
  const __bf16* __restrict__ Bt = args.Wt[mode];
  const float*  __restrict__ bias = args.bias[mode];

  __shared__ __attribute__((aligned(32))) __bf16 As[2][BM * BK];
  __shared__ __attribute__((aligned(32))) __bf16 Bs[2][BN * BK];

  const int tid = threadIdx.x, lane = tid & 31, wid = tid >> 5;
  const int lm = lane & 15, lh = lane >> 4;
  const int wm = wid >> 2, wn = wid & 3;          // 2x4 wave grid: 64x32/wave
  const int rowBase = blockIdx.x * BM;
  const int colBase = blockIdx.y * BN;

  // A tile: per-lane async copies. B tile: one TDM 2D descriptor from wave 0.
  auto load_tiles = [&](int buf, int kt) {
#pragma unroll
    for (int it = 0; it < 4; ++it) {
      int c = tid + it * 256, r = c >> 3, c8 = (c & 7) * 8;
      async_copy16(&As[buf][r * BK + c8],
                   &A[(size_t)(rowBase + r) * 1024 + kt + c8]);
#if !HAVE_TDM
      async_copy16(&Bs[buf][r * BK + c8],
                   &Bt[(size_t)(colBase + r) * 1024 + kt + c8]);
#endif
    }
#if HAVE_TDM
    if (wid == 0)
      tdm_load_2d(&Bs[buf][0], &Bt[(size_t)colBase * 1024 + kt],
                  /*tensorD0=*/1024, /*tensorD1=*/1024,
                  /*tileD0=*/BK, /*tileD1=*/BN, /*strideD0=*/1024);
#endif
  };
  auto sync_tiles = [&]() {
    wait_async0();
#if HAVE_TDM
    if (wid == 0) wait_tensor0();
#endif
    __syncthreads();
  };

  v8f acc[4][2] = {};

  load_tiles(0, 0);                               // prologue: tile 0
  sync_tiles();

  int cur = 0;
  for (int kt = 0; kt < 1024; kt += BK) {
    if (kt + BK < 1024) load_tiles(cur ^ 1, kt + BK);  // prefetch next tile

#pragma unroll
    for (int k0 = 0; k0 < BK; k0 += 32) {
      v16bf bfr[2], afr[4];
#pragma unroll
      for (int nn = 0; nn < 2; ++nn)
        bfr[nn] = frag_b(&Bs[cur][(wn * 32 + nn * 16 + lm) * BK], k0, lh);
#pragma unroll
      for (int mm = 0; mm < 4; ++mm)
        afr[mm] = frag_a(&As[cur][(wm * 64 + mm * 16 + lm) * BK], k0, lh);
#pragma unroll
      for (int mm = 0; mm < 4; ++mm)
#pragma unroll
        for (int nn = 0; nn < 2; ++nn)
          acc[mm][nn] = wmma_bf16(afr[mm], bfr[nn], acc[mm][nn]);
    }

    sync_tiles();                                 // next tile resident
    cur ^= 1;
  }

  // Epilogue: bias + (mode-dependent) RoPE / transposed stores.
#pragma unroll
  for (int mm = 0; mm < 4; ++mm)
#pragma unroll
    for (int nn = 0; nn < 2; ++nn)
#pragma unroll
      for (int v = 0; v < 8; ++v) {
        const int rG = rowBase + wm * 64 + mm * 16 + lh * 8 + v;
        const int cG = colBase + wn * 32 + nn * 16 + lm;
        float self = acc[mm][nn][v] + bias[cG];
        if (mode == 3) { args.out_o_f32[(size_t)rG * 1024 + cG] = self; continue; }
        const int i = rG >> 2, b = rG & 3;        // rows are (i*BSZ + b)
        const int n = cG >> 6, d = cG & 63;
        const size_t head = (size_t)(b * NH + n);
        if (mode == 1) args.out_k_f32[(size_t)rG * 1024 + cG] = self;
        if (mode == 2) {
          args.out_v_f32[(size_t)rG * 1024 + cG] = self;
          args.vt[(head * DH + d) * QLEN + i] = (__bf16)self;   // V transposed
          continue;
        }
        // RoPE: the (even,odd) pair sits in lanes (lane, lane^1).
        const float cv = args.fcos[i * 32 + (d >> 1)];
        const float sv = args.fsin[i * 32 + (d >> 1)];
        const float partner = __shfl_xor(self, 1, 32);
        const float ro = self * cv + ((d & 1) ? partner * sv : -partner * sv);
        __bf16* dst = (mode == 0) ? args.qro : args.kro;
        dst[(head * QLEN + i) * DH + d] = (__bf16)ro;
      }
}

// --------------------------- flash attention kernel ------------------------
__global__ __launch_bounds__(256) void attn_kernel(
    const __bf16* __restrict__ Qro, const __bf16* __restrict__ Kro,
    const __bf16* __restrict__ Vt, __bf16* __restrict__ Ob) {
  const int head = blockIdx.x;                    // b*NH + n
  const int qt   = blockIdx.y;                    // 128-query tile
  const int b = head >> 4, n = head & 15;
  const __bf16* __restrict__ Qh = Qro + (size_t)head * QLEN * DH;
  const __bf16* __restrict__ Kh = Kro + (size_t)head * QLEN * DH;
  const __bf16* __restrict__ Vh = Vt  + (size_t)head * DH * QLEN;

  __shared__ __attribute__((aligned(32))) __bf16 Qs[128 * 64];
  __shared__ __attribute__((aligned(32))) __bf16 Ks[2][64 * 64];
  __shared__ __attribute__((aligned(32))) __bf16 Vs[2][64 * 64];   // [d][j]
  __shared__ __attribute__((aligned(32))) __bf16 Ps[128 * 64];

  const int tid = threadIdx.x, lane = tid & 31, wid = tid >> 5;
  const int lm = lane & 15, lh = lane >> 4;
  const int m0 = wid * 16;                        // wave owns 16 query rows

  // K slab: per-lane async. V^T slab (strided 2D): TDM descriptor, wave 0.
  auto load_kv = [&](int buf, int kt) {
#pragma unroll
    for (int it = 0; it < 2; ++it) {
      int c = tid + it * 256, r = c >> 3, c8 = (c & 7) * 8;
      async_copy16(&Ks[buf][r * 64 + c8],
                   &Kh[(size_t)(kt + r) * 64 + c8]);
#if !HAVE_TDM
      async_copy16(&Vs[buf][r * 64 + c8],
                   &Vh[(size_t)r * QLEN + kt + c8]);
#endif
    }
#if HAVE_TDM
    if (wid == 0)
      tdm_load_2d(&Vs[buf][0], &Vh[kt],
                  /*tensorD0=*/QLEN, /*tensorD1=*/DH,
                  /*tileD0=*/64, /*tileD1=*/64, /*strideD0=*/QLEN);
#endif
  };
  auto sync_tiles = [&]() {
    wait_async0();
#if HAVE_TDM
    if (wid == 0) wait_tensor0();
#endif
    __syncthreads();
  };

  // prologue: Q tile (128x64 contiguous) + first K/V slab
#pragma unroll
  for (int it = 0; it < 4; ++it) {
    int c = tid + it * 256, r = c >> 3, c8 = (c & 7) * 8;
    async_copy16(&Qs[r * 64 + c8],
                 &Qh[(size_t)(qt * 128 + r) * 64 + c8]);
  }
  load_kv(0, 0);
  sync_tiles();

  v8f   acc[4] = {};
  float mrow[8], lrow[8];
#pragma unroll
  for (int v = 0; v < 8; ++v) { mrow[v] = -1e30f; lrow[v] = 0.f; }

  int cur = 0;
  for (int kt = 0; kt < QLEN; kt += 64) {
    if (kt + 64 < QLEN) load_kv(cur ^ 1, kt + 64);     // prefetch next slab

    // S = Q K^T (16 rows x 64 keys per wave), 8 WMMAs
    v8f s[4] = {};
    {
      v16bf aq0 = frag_a(&Qs[(m0 + lm) * 64], 0, lh);
      v16bf aq1 = frag_a(&Qs[(m0 + lm) * 64], 32, lh);
#pragma unroll
      for (int f = 0; f < 4; ++f) {
        v16bf bk0 = frag_b(&Ks[cur][(f * 16 + lm) * 64], 0, lh);
        v16bf bk1 = frag_b(&Ks[cur][(f * 16 + lm) * 64], 32, lh);
        s[f] = wmma_bf16(aq0, bk0, s[f]);
        s[f] = wmma_bf16(aq1, bk1, s[f]);
      }
    }

    // Online softmax over this 64-key slab.
    float rm[8];
#pragma unroll
    for (int v = 0; v < 8; ++v) rm[v] = -1e30f;
#pragma unroll
    for (int f = 0; f < 4; ++f)
#pragma unroll
      for (int v = 0; v < 8; ++v) {
        s[f][v] *= SCALE;
        rm[v] = fmaxf(rm[v], s[f][v]);
      }
#pragma unroll
    for (int v = 0; v < 8; ++v) {                 // row-max across 16 lanes
      rm[v] = fmaxf(rm[v], __shfl_xor(rm[v], 1, 32));
      rm[v] = fmaxf(rm[v], __shfl_xor(rm[v], 2, 32));
      rm[v] = fmaxf(rm[v], __shfl_xor(rm[v], 4, 32));
      rm[v] = fmaxf(rm[v], __shfl_xor(rm[v], 8, 32));
    }
    float alpha[8], rs[8];
#pragma unroll
    for (int v = 0; v < 8; ++v) {
      float mn = fmaxf(mrow[v], rm[v]);
      alpha[v] = __expf(mrow[v] - mn);
      mrow[v] = mn;
      rs[v] = 0.f;
    }
#pragma unroll
    for (int f = 0; f < 4; ++f)
#pragma unroll
      for (int v = 0; v < 8; ++v) {
        float p = __expf(s[f][v] - mrow[v]);
        rs[v] += p;
        Ps[(m0 + lh * 8 + v) * 64 + f * 16 + lm] = (__bf16)p;
      }
#pragma unroll
    for (int v = 0; v < 8; ++v) {
      rs[v] += __shfl_xor(rs[v], 1, 32);
      rs[v] += __shfl_xor(rs[v], 2, 32);
      rs[v] += __shfl_xor(rs[v], 4, 32);
      rs[v] += __shfl_xor(rs[v], 8, 32);
      lrow[v] = lrow[v] * alpha[v] + rs[v];
    }
#pragma unroll
    for (int f = 0; f < 4; ++f)
#pragma unroll
      for (int v = 0; v < 8; ++v) acc[f][v] *= alpha[v];

    asm volatile("s_wait_dscnt 0" ::: "memory");  // P stores visible to own wave

    // O += P V  (P rows are wave-private in LDS), 8 WMMAs
    {
      v16bf ap0 = frag_a(&Ps[(m0 + lm) * 64], 0, lh);
      v16bf ap1 = frag_a(&Ps[(m0 + lm) * 64], 32, lh);
#pragma unroll
      for (int f = 0; f < 4; ++f) {
        v16bf bv0 = frag_b(&Vs[cur][(f * 16 + lm) * 64], 0, lh);
        v16bf bv1 = frag_b(&Vs[cur][(f * 16 + lm) * 64], 32, lh);
        acc[f] = wmma_bf16(ap0, bv0, acc[f]);
        acc[f] = wmma_bf16(ap1, bv1, acc[f]);
      }
    }

    sync_tiles();                                 // next slab resident
    cur ^= 1;
  }

  // Normalize and emit bf16 rows for the output projection: row (i*BSZ+b).
#pragma unroll
  for (int f = 0; f < 4; ++f)
#pragma unroll
    for (int v = 0; v < 8; ++v) {
      const int qi = qt * 128 + m0 + lh * 8 + v;
      const int d  = f * 16 + lm;
      const float val = acc[f][v] / lrow[v];
      Ob[((size_t)qi * BSZ + b) * H + n * DH + d] = (__bf16)val;
    }
}

// ------------------------------- launch glue -------------------------------
extern "C" void kernel_launch(void* const* d_in, const int* in_sizes, int n_in,
                              void* d_out, int out_size, void* d_ws, size_t ws_size,
                              hipStream_t stream) {
  const float* x    = (const float*)d_in[0];
  const float* fcos = (const float*)d_in[1];
  const float* fsin = (const float*)d_in[2];
  const float* Wq = (const float*)d_in[3];  const float* bq = (const float*)d_in[4];
  const float* Wk = (const float*)d_in[5];  const float* bk = (const float*)d_in[6];
  const float* Wv = (const float*)d_in[7];  const float* bv = (const float*)d_in[8];
  const float* Wo = (const float*)d_in[9];  const float* bo = (const float*)d_in[10];
  float* out = (float*)d_out;

  // Workspace carve-up (bf16 staging, ~88 MB total).
  char* ws = (char*)d_ws;
  __bf16* xb  = (__bf16*)ws;                          ws += (size_t)ROWS * DM * 2;
  __bf16* wtq = (__bf16*)ws;                          ws += (size_t)DM * H * 2;
  __bf16* wtk = (__bf16*)ws;                          ws += (size_t)DM * H * 2;
  __bf16* wtv = (__bf16*)ws;                          ws += (size_t)DM * H * 2;
  __bf16* wto = (__bf16*)ws;                          ws += (size_t)H * DM * 2;
  __bf16* qro = (__bf16*)ws;                          ws += (size_t)BSZ * NH * QLEN * DH * 2;
  __bf16* kro = (__bf16*)ws;                          ws += (size_t)BSZ * NH * QLEN * DH * 2;
  __bf16* vt  = (__bf16*)ws;                          ws += (size_t)BSZ * NH * DH * QLEN * 2;
  __bf16* ob  = (__bf16*)ws;                          ws += (size_t)ROWS * H * 2;

  // 1) conversions
  cvt_x<<<(ROWS * DM) / (256 * 4), 256, 0, stream>>>(x, xb);
  cvt_w_t<<<(DM * H) / 256, 256, 0, stream>>>(Wq, wtq);
  cvt_w_t<<<(DM * H) / 256, 256, 0, stream>>>(Wk, wtk);
  cvt_w_t<<<(DM * H) / 256, 256, 0, stream>>>(Wv, wtv);
  cvt_w_t<<<(H * DM) / 256, 256, 0, stream>>>(Wo, wto);

  GemmArgs ga;
  ga.A = xb;
  ga.Wt[0] = wtq; ga.Wt[1] = wtk; ga.Wt[2] = wtv; ga.Wt[3] = wto;
  ga.bias[0] = bq; ga.bias[1] = bk; ga.bias[2] = bv; ga.bias[3] = bo;
  ga.fcos = fcos; ga.fsin = fsin;
  ga.out_o_f32 = out;
  ga.out_k_f32 = out + (size_t)ROWS * DM;
  ga.out_v_f32 = out + (size_t)2 * ROWS * DM;
  ga.qro = qro; ga.kro = kro; ga.vt = vt;

  // 2) QKV projections (z = 0:Q, 1:K, 2:V) with fused RoPE/transpose epilogue
  ga.modeBase = 0;
  gemm_qkvo<<<dim3(ROWS / BM, H / BN, 3), 256, 0, stream>>>(ga);

  // 3) flash attention: 64 heads x 16 query tiles
  attn_kernel<<<dim3(BSZ * NH, QLEN / 128), 256, 0, stream>>>(qro, kro, vt, ob);

  // 4) output projection
  GemmArgs go = ga;
  go.A = ob;
  go.modeBase = 3;
  gemm_qkvo<<<dim3(ROWS / BM, DM / BN, 1), 256, 0, stream>>>(go);
}
